// LocalSelfAttentionDecoderLayer_4020089389258
// MI455X (gfx1250) — compile-verified, matrix operand order checked by reading, not run on hardware
//
#include <hip/hip_runtime.h>
#include <stdint.h>

#define B_  8
#define NT_ 1024
#define NM_ 2048
#define C_  1024
#define H_  8
#define D_  128
#define FF_ 4096

typedef unsigned short u16;
typedef __bf16  v16bf __attribute__((ext_vector_type(16)));
typedef float   v8f   __attribute__((ext_vector_type(8)));

struct Frag32B { uint4 lo, hi; };

__device__ __forceinline__ u16 f2bf(float f) {
  uint32_t u = __builtin_bit_cast(uint32_t, f);
  u += 0x7FFFu + ((u >> 16) & 1u);              // round-to-nearest-even
  return (u16)(u >> 16);
}

// A/B fragment for v_wmma_f32_16x16x32_bf16:
// lane l<16 holds k = {k0..k0+7, k0+16..k0+23} with k0=0; lanes 16-31 use k0=8.
__device__ __forceinline__ v16bf load_frag(const u16* base, int k0) {
  Frag32B f;
  f.lo = *(const uint4*)(base + k0);
  f.hi = *(const uint4*)(base + k0 + 16);
  return __builtin_bit_cast(v16bf, f);
}

__device__ __forceinline__ v8f wmma_bf16(v16bf a, v16bf b, v8f c) {
  return __builtin_amdgcn_wmma_f32_16x16x32_bf16(false, a, false, b, (short)0, c, false, false);
}

__device__ __forceinline__ v8f zero8() {
  v8f z = {0.f,0.f,0.f,0.f,0.f,0.f,0.f,0.f};
  return z;
}

// async DMA: 16 bytes per lane, global -> LDS, tracked by ASYNCcnt (gfx1250)
__device__ __forceinline__ void async_copy16(uint32_t lds_byte_off, const void* gptr) {
  asm volatile("global_load_async_to_lds_b128 %0, %1, off"
               :: "v"(lds_byte_off), "v"((uint64_t)(uintptr_t)gptr)
               : "memory");
}
__device__ __forceinline__ void wait_async0() {
  asm volatile("s_wait_asynccnt 0x0" ::: "memory");
}

// ---------------- f32 -> bf16 conversion ----------------
__global__ __launch_bounds__(256) void cvt_k(const float* __restrict__ in,
                                             u16* __restrict__ out, long n) {
  long i = ((long)blockIdx.x * 256 + threadIdx.x) * 4;
  if (i >= n) return;
  float4 f = *(const float4*)(in + i);
  out[i+0] = f2bf(f.x); out[i+1] = f2bf(f.y);
  out[i+2] = f2bf(f.z); out[i+3] = f2bf(f.w);
}

// ---------------- generic GEMM: out = act(A[MxK] @ W[NxK]^T + bias) (+resid) ----------------
// block tile 128x128, 8 waves as 4(M) x 2(N); wave tile 32x64 = 2x4 wmma accumulators.
// Tiles staged via GLOBAL_LOAD_ASYNC_TO_LDS_B128 (no VGPR round trip).
#define LDSA_STRIDE 40                  // 32-k tile + 8 pad (u16 elements)
#define LDSB_BASE   (128 * LDSA_STRIDE) // element offset of W tile

template<bool RELU>
__global__ __launch_bounds__(256)
void gemm_bf16_k(const u16* __restrict__ A, const u16* __restrict__ W,
                 const float* __restrict__ bias, const float* __restrict__ resid,
                 float* __restrict__ outF, u16* __restrict__ outB,
                 int M, int N, int K)
{
  extern __shared__ u16 smem[];         // dynamic LDS: offsets start at 0
  u16* As = smem;                       // [128][40]
  u16* Bs = smem + LDSB_BASE;           // [128][40]

  const int tid  = threadIdx.x;
  const int lane = tid & 31, wave = tid >> 5;
  const int l16 = lane & 15, half = lane >> 4, koff = half * 8;
  const int wm = wave & 3, wn = wave >> 2;
  const int m0 = blockIdx.y * 128, n0 = blockIdx.x * 128;

  v8f acc[2][4];
  #pragma unroll
  for (int i = 0; i < 2; ++i)
    #pragma unroll
    for (int j = 0; j < 4; ++j) acc[i][j] = zero8();

  const int nk = K >> 5;
  for (int kb = 0; kb < nk; ++kb) {
    __syncthreads();                    // previous tile fully consumed
    for (int c = tid; c < 512; c += 256) {
      const int row = c >> 2, col = (c & 3) * 8;
      const uint32_t loff = (uint32_t)((row * LDSA_STRIDE + col) * 2);
      async_copy16(loff,
                   A + (size_t)(m0 + row) * K + kb * 32 + col);
      async_copy16(loff + (uint32_t)(LDSB_BASE * 2),
                   W + (size_t)(n0 + row) * K + kb * 32 + col);
    }
    if (kb + 1 < nk) {  // hint next tiles toward L2
      __builtin_prefetch(W + (size_t)(n0 + (tid >> 1)) * K + (kb + 1) * 32 + (tid & 1) * 16, 0, 1);
      __builtin_prefetch(A + (size_t)(m0 + (tid >> 1)) * K + (kb + 1) * 32 + (tid & 1) * 16, 0, 1);
    }
    wait_async0();                      // this wave's DMAs landed in LDS
    __syncthreads();                    // all waves' DMAs visible

    v16bf af[2], bw[4];
    #pragma unroll
    for (int i = 0; i < 2; ++i)
      af[i] = load_frag(As + (wm * 32 + i * 16 + l16) * LDSA_STRIDE, koff);
    #pragma unroll
    for (int j = 0; j < 4; ++j)
      bw[j] = load_frag(Bs + (wn * 64 + j * 16 + l16) * LDSA_STRIDE, koff);
    #pragma unroll
    for (int i = 0; i < 2; ++i)
      #pragma unroll
      for (int j = 0; j < 4; ++j)
        acc[i][j] = wmma_bf16(af[i], bw[j], acc[i][j]);
  }

  #pragma unroll
  for (int i = 0; i < 2; ++i) {
    #pragma unroll
    for (int j = 0; j < 4; ++j) {
      const int n = n0 + wn * 64 + j * 16 + l16;
      const float bv = bias ? bias[n] : 0.f;
      #pragma unroll
      for (int r = 0; r < 8; ++r) {
        const int m = m0 + wm * 32 + i * 16 + half * 8 + r;   // C-matrix layout
        float v = acc[i][j][r] + bv;
        if (RELU) v = fmaxf(v, 0.f);
        const size_t idx = (size_t)m * N + n;
        if (resid) v += resid[idx];
        if (outF)  outF[idx] = v;
        if (outB)  outB[idx] = f2bf(v);
      }
    }
  }
}

// ---------------- LayerNorm (one block per row of C_=1024) ----------------
__global__ __launch_bounds__(256)
void ln_k(const float* __restrict__ x, const float* __restrict__ g,
          const float* __restrict__ be, float* __restrict__ outF,
          u16* __restrict__ outB)
{
  const int row = blockIdx.x;
  const float* xr = x + (size_t)row * C_;
  float s = 0.f, s2 = 0.f;
  for (int i = threadIdx.x; i < C_; i += 256) { float v = xr[i]; s += v; s2 += v * v; }
  #pragma unroll
  for (int off = 16; off > 0; off >>= 1) {
    s  += __shfl_xor(s,  off, 32);
    s2 += __shfl_xor(s2, off, 32);
  }
  __shared__ float sh0[8], sh1[8];
  if ((threadIdx.x & 31) == 0) { sh0[threadIdx.x >> 5] = s; sh1[threadIdx.x >> 5] = s2; }
  __syncthreads();
  s = 0.f; s2 = 0.f;
  #pragma unroll
  for (int w = 0; w < 8; ++w) { s += sh0[w]; s2 += sh1[w]; }
  const float mean = s * (1.f / C_);
  const float var  = s2 * (1.f / C_) - mean * mean;
  const float inv  = rsqrtf(var + 1e-5f);
  for (int i = threadIdx.x; i < C_; i += 256) {
    float v = (xr[i] - mean) * inv * g[i] + be[i];
    if (outF) outF[(size_t)row * C_ + i] = v;
    if (outB) outB[(size_t)row * C_ + i] = f2bf(v);
  }
}

// ---------------- flash cross-attention: 8 heads, d=128, M=2048 keys ----------------
// grid (NT/128, H, B); 8 waves, each owns 16 query rows; key blocks of 64.
__global__ __launch_bounds__(256)
void flash_k(const u16* __restrict__ Q, const u16* __restrict__ Kb,
             const u16* __restrict__ Vb, u16* __restrict__ O)
{
  __shared__ u16 Ks[64][136];      // keys x d (row-major, padded)
  __shared__ u16 Vt[128][72];      // d x keys (transposed during staging)
  __shared__ u16 Ps[8][16][72];    // per-wave P tile (16 rows x 64 keys)
  const int tid  = threadIdx.x;
  const int lane = tid & 31, wave = tid >> 5;
  const int l16 = lane & 15, half = lane >> 4, koff = half * 8;
  const int bi = blockIdx.z, hi = blockIdx.y;
  const int q0 = blockIdx.x * 128 + wave * 16;
  const float scale = 0.08838834764831845f;   // 1/sqrt(d)

  // preload this wave's Q rows directly into A-fragments (d = 4 k-steps of 32)
  const u16* qr = Q + ((size_t)(bi * NT_) + q0 + l16) * C_ + hi * D_;
  v16bf aq[4];
  #pragma unroll
  for (int ks = 0; ks < 4; ++ks) aq[ks] = load_frag(qr, ks * 32 + koff);

  v8f o[8];
  #pragma unroll
  for (int nt = 0; nt < 8; ++nt) o[nt] = zero8();
  float mrun[8], srun[8];
  #pragma unroll
  for (int r = 0; r < 8; ++r) { mrun[r] = -1e30f; srun[r] = 0.f; }

  const u16* Kg0 = Kb + (size_t)(bi * NM_) * C_ + hi * D_;
  const u16* Vg0 = Vb + (size_t)(bi * NM_) * C_ + hi * D_;

  for (int kb = 0; kb < NM_ / 64; ++kb) {
    __syncthreads();
    const u16* Kg = Kg0 + (size_t)kb * 64 * C_;
    const u16* Vg = Vg0 + (size_t)kb * 64 * C_;
    #pragma unroll
    for (int i = 0; i < 4; ++i) {
      int c = tid + i * 256;
      int row = c >> 4, col = (c & 15) * 8;
      *(uint4*)&Ks[row][col] = *(const uint4*)(Kg + (size_t)row * C_ + col);
      union { uint4 v4; u16 e[8]; } u;
      u.v4 = *(const uint4*)(Vg + (size_t)row * C_ + col);
      #pragma unroll
      for (int e = 0; e < 8; ++e) Vt[col + e][row] = u.e[e];  // transpose into LDS
    }
    __syncthreads();

    // S = Q @ K^T  (4 key tiles of 16, 4 k-steps over d)
    v8f s[4];
    #pragma unroll
    for (int nt = 0; nt < 4; ++nt) s[nt] = zero8();
    #pragma unroll
    for (int ks = 0; ks < 4; ++ks)
      #pragma unroll
      for (int nt = 0; nt < 4; ++nt) {
        v16bf bk = load_frag(&Ks[nt * 16 + l16][0], ks * 32 + koff);
        s[nt] = wmma_bf16(aq[ks], bk, s[nt]);
      }

    // online softmax per row (row = half*8 + r; cols live across the 16 lanes of the half)
    #pragma unroll
    for (int r = 0; r < 8; ++r) {
      float tm = -1e30f;
      #pragma unroll
      for (int nt = 0; nt < 4; ++nt) tm = fmaxf(tm, s[nt][r] * scale);
      #pragma unroll
      for (int off = 1; off < 16; off <<= 1) tm = fmaxf(tm, __shfl_xor(tm, off, 32));
      const float mnew  = fmaxf(mrun[r], tm);
      const float alpha = __expf(mrun[r] - mnew);
      mrun[r] = mnew;
      float rs = 0.f;
      #pragma unroll
      for (int nt = 0; nt < 4; ++nt) {
        float p = __expf(s[nt][r] * scale - mnew);
        rs += p;
        Ps[wave][half * 8 + r][nt * 16 + l16] = f2bf(p);
      }
      #pragma unroll
      for (int off = 1; off < 16; off <<= 1) rs += __shfl_xor(rs, off, 32);
      srun[r] = srun[r] * alpha + rs;
      #pragma unroll
      for (int nt = 0; nt < 8; ++nt) o[nt][r] = o[nt][r] * alpha;
    }

    // O += P @ V  (2 k-steps over 64 keys, 8 d-tiles of 16)
    #pragma unroll
    for (int kc = 0; kc < 2; ++kc) {
      v16bf ap = load_frag(&Ps[wave][l16][0], kc * 32 + koff);
      #pragma unroll
      for (int nt = 0; nt < 8; ++nt) {
        v16bf bv = load_frag(&Vt[nt * 16 + l16][0], kc * 32 + koff);
        o[nt] = wmma_bf16(ap, bv, o[nt]);
      }
    }
  }

  u16* og = O + ((size_t)(bi * NT_) + q0) * C_ + hi * D_;
  #pragma unroll
  for (int r = 0; r < 8; ++r) {
    const float rinv = 1.f / srun[r];
    #pragma unroll
    for (int nt = 0; nt < 8; ++nt)
      og[(size_t)(half * 8 + r) * C_ + nt * 16 + l16] = f2bf(o[nt][r] * rinv);
  }
}

// ---------------- host orchestration ----------------
extern "C" void kernel_launch(void* const* d_in, const int* in_sizes, int n_in,
                              void* d_out, int out_size, void* d_ws, size_t ws_size,
                              hipStream_t stream)
{
  const float* tgt       = (const float*)d_in[0];
  const float* mem       = (const float*)d_in[1];
  const float* qkv_w     = (const float*)d_in[2];
  const float* qkv_b     = (const float*)d_in[3];
  const float* sa_out_w  = (const float*)d_in[4];
  const float* sa_out_b  = (const float*)d_in[5];
  const float* in_proj_w = (const float*)d_in[6];
  const float* in_proj_b = (const float*)d_in[7];
  const float* mha_out_w = (const float*)d_in[8];
  const float* mha_out_b = (const float*)d_in[9];
  const float* lin1_w    = (const float*)d_in[10];
  const float* lin1_b    = (const float*)d_in[11];
  const float* lin2_w    = (const float*)d_in[12];
  const float* lin2_b    = (const float*)d_in[13];
  const float* ln1_g     = (const float*)d_in[14];
  const float* ln1_b     = (const float*)d_in[15];
  const float* ln2_g     = (const float*)d_in[16];
  const float* ln2_b     = (const float*)d_in[17];
  const float* ln3_g     = (const float*)d_in[18];
  const float* ln3_b     = (const float*)d_in[19];
  (void)in_sizes; (void)n_in; (void)out_size; (void)ws_size;

  char* ws = (char*)d_ws;
  size_t off = 0;
  auto alloc = [&](size_t bytes) -> void* {
    void* p = ws + off;
    off = (off + bytes + 255) & ~(size_t)255;
    return p;
  };

  const size_t MT = (size_t)B_ * NT_;   // 8192 target rows
  const size_t MM = (size_t)B_ * NM_;   // 16384 memory rows

  u16*   tgtB  = (u16*)alloc(MT * C_ * 2);
  u16*   memB  = (u16*)alloc(MM * C_ * 2);
  u16*   wvsaB = (u16*)alloc((size_t)C_ * C_ * 2);
  u16*   saowB = (u16*)alloc((size_t)C_ * C_ * 2);
  u16*   inpB  = (u16*)alloc((size_t)3 * C_ * C_ * 2);
  u16*   mhowB = (u16*)alloc((size_t)C_ * C_ * 2);
  u16*   l1B   = (u16*)alloc((size_t)FF_ * C_ * 2);
  u16*   l2B   = (u16*)alloc((size_t)C_ * FF_ * 2);
  u16*   vsaB  = (u16*)alloc(MT * C_ * 2);
  float* x0F   = (float*)alloc(MT * C_ * 4);
  float* x1F   = (float*)alloc(MT * C_ * 4);
  u16*   x1B   = (u16*)alloc(MT * C_ * 2);
  u16*   qB    = (u16*)alloc(MT * C_ * 2);
  u16*   kB    = (u16*)alloc(MM * C_ * 2);
  u16*   vB    = (u16*)alloc(MM * C_ * 2);
  u16*   caB   = (u16*)alloc(MT * C_ * 2);
  float* x2p   = (float*)alloc(MT * C_ * 4);
  float* x2F   = (float*)alloc(MT * C_ * 4);
  u16*   x2B   = (u16*)alloc(MT * C_ * 2);
  u16*   h1B   = (u16*)alloc(MT * FF_ * 2);
  float* ffp   = (float*)alloc(MT * C_ * 4);

  auto cvt = [&](const float* src, u16* dst, size_t n) {
    cvt_k<<<dim3((unsigned)(n / 4 / 256)), 256, 0, stream>>>(src, dst, (long)n);
  };
  cvt(tgt, tgtB, MT * C_);
  cvt(mem, memB, MM * C_);
  cvt(qkv_w + (size_t)2 * C_ * C_, wvsaB, (size_t)C_ * C_);   // only v-slice needed
  cvt(sa_out_w, saowB, (size_t)C_ * C_);
  cvt(in_proj_w, inpB, (size_t)3 * C_ * C_);
  cvt(mha_out_w, mhowB, (size_t)C_ * C_);
  cvt(lin1_w, l1B, (size_t)FF_ * C_);
  cvt(lin2_w, l2B, (size_t)C_ * FF_);

  const size_t gemm_lds = (size_t)2 * 128 * LDSA_STRIDE * 2;  // 20480 B
  auto gemm = [&](const u16* A, const u16* W, const float* bias, const float* resid,
                  float* oF, u16* oB, int M, int N, int K, bool relu) {
    dim3 g(N / 128, M / 128);
    if (relu) gemm_bf16_k<true ><<<g, 256, gemm_lds, stream>>>(A, W, bias, resid, oF, oB, M, N, K);
    else      gemm_bf16_k<false><<<g, 256, gemm_lds, stream>>>(A, W, bias, resid, oF, oB, M, N, K);
  };

  // --- self-attention degenerates to the v-projection (diagonal mask => identity attn) ---
  gemm(tgtB, wvsaB, qkv_b + 2 * C_, nullptr, nullptr, vsaB, (int)MT, C_, C_, false);
  gemm(vsaB, saowB, sa_out_b, tgt, x0F, nullptr, (int)MT, C_, C_, false);  // +tgt residual
  ln_k<<<(unsigned)MT, 256, 0, stream>>>(x0F, ln1_g, ln1_b, x1F, x1B);

  // --- cross-attention projections ---
  gemm(x1B,  inpB,                          in_proj_b,          nullptr, nullptr, qB, (int)MT, C_, C_, false);
  gemm(memB, inpB + (size_t)C_ * C_,        in_proj_b + C_,     nullptr, nullptr, kB, (int)MM, C_, C_, false);
  gemm(memB, inpB + (size_t)2 * C_ * C_,    in_proj_b + 2 * C_, nullptr, nullptr, vB, (int)MM, C_, C_, false);

  flash_k<<<dim3(NT_ / 128, H_, B_), 256, 0, stream>>>(qB, kB, vB, caB);

  gemm(caB, mhowB, mha_out_b, x1F, x2p, nullptr, (int)MT, C_, C_, false);  // +x1 residual
  ln_k<<<(unsigned)MT, 256, 0, stream>>>(x2p, ln2_g, ln2_b, x2F, x2B);

  // --- FFN ---
  gemm(x2B, l1B, lin1_b, nullptr, nullptr, h1B, (int)MT, FF_, C_, true);   // ReLU
  gemm(h1B, l2B, lin2_b, x2F, ffp, nullptr, (int)MT, C_, FF_, false);      // +x2 residual
  ln_k<<<(unsigned)MT, 256, 0, stream>>>(ffp, ln3_g, ln3_b, (float*)d_out, nullptr);
}
